// KDEformer_90675349553509
// MI455X (gfx1250) — matrix operand order repaired and checked by reading, not compiled, not run on hardware
//
#include <hip/hip_runtime.h>
#include <hip/hip_bf16.h>
#include <math.h>

// ---------------- constants ----------------
constexpr int kB = 2, kH = 8, kN = 4096, kD = 64;
constexpr int kBH = kB * kH;               // 16
constexpr int kNPROJ = 7;
constexpr int kBUCKET = 64;
constexpr int kNBLK = kN / kBUCKET;        // 64
constexpr int kSAMPLE = 800;
constexpr int kSPAD = 832;                 // 13 chunks of 64
constexpr int kDV = 65;                    // d + 1
constexpr int kDVP = 80;                   // padded to 5 N-tiles of 16
constexpr int kATTLD = 66;                 // att row stride (floats)

typedef __attribute__((ext_vector_type(16))) _Float16 v16h;
typedef __attribute__((ext_vector_type(8)))  float    v8f;
typedef __attribute__((ext_vector_type(4)))  unsigned u32x4;
typedef __attribute__((ext_vector_type(8)))  int      i32x8;
typedef __attribute__((ext_vector_type(4)))  int      i32x4;
typedef _Float16 h16;

// ---------------- RNG helpers ----------------
__device__ __forceinline__ unsigned pcg(unsigned v) {
  unsigned s = v * 747796405u + 2891336453u;
  unsigned w = ((s >> ((s >> 28u) + 4u)) ^ s) * 277803737u;
  return (w >> 22u) ^ w;
}
__device__ __forceinline__ float rng_normal(unsigned seed) {
  unsigned a  = pcg(seed * 2654435761u + 0x9E3779B9u);
  unsigned b2 = pcg(a + 0x85EBCA6Bu);
  float u1 = ((float)a  + 0.5f) * (1.0f / 4294967296.0f);
  float u2 = ((float)b2 + 0.5f) * (1.0f / 4294967296.0f);
  return sqrtf(-2.0f * __logf(u1)) * __cosf(6.28318530718f * u2);
}

// ---------------- Tensor Data Mover: contiguous 1-D tile -> LDS ----------------
// Builds a D# per CDNA5 ISA: group0 = {flags, lds_addr,
// global_addr, type=2}; group1 = {data_size=8B, tensor_dim0, tensor_dim1=1,
// tile_dim0=nelem8, strides}; groups 2/3 zero (<=2D tensor).
__device__ __forceinline__ void tdm_load_1d(unsigned lds_byte_addr,
                                            const void* gptr, unsigned nelem8) {
  unsigned long long ga = (unsigned long long)(uintptr_t)gptr;
  u32x4 g0;
  g0[0] = 1u;                                            // count=1 (valid), user mode
  g0[1] = lds_byte_addr;                                 // LDS destination (bytes)
  g0[2] = (unsigned)(ga & 0xFFFFFFFFull);                // global_addr[31:0]
  g0[3] = (unsigned)((ga >> 32) & 0x01FFFFFFull)         // global_addr[56:32]
        | (2u << 30);                                    // type = 2 ("image")
  i32x8 g1;
  g1[0] = (int)(3u << 16);                               // wg_mask=0, data_size=3 (8B)
  g1[1] = (int)((nelem8 & 0xFFFFu) << 16);               // tensor_dim0[15:0]
  g1[2] = (int)(((nelem8 >> 16) & 0xFFFFu) | (1u << 16)); // tensor_dim0[31:16] | tensor_dim1=1
  g1[3] = (int)((nelem8 & 0xFFFFu) << 16);               // tile_dim0 = nelem8
  g1[4] = 0;                                             // tile_dim1=0 (1-D), tile_dim2=0
  g1[5] = (int)nelem8;                                   // tensor_dim0_stride[31:0]
  g1[6] = 0;
  g1[7] = 0;
  i32x4 gz = {0, 0, 0, 0};
#if defined(__clang_major__) && (__clang_major__ >= 23)
  i32x8 gz8 = {0, 0, 0, 0, 0, 0, 0, 0};
  __builtin_amdgcn_tensor_load_to_lds(g0, g1, gz, gz, gz8, 0);
#else
  __builtin_amdgcn_tensor_load_to_lds(g0, g1, gz, gz, 0);
#endif
}
__device__ __forceinline__ unsigned lds_off(const void* p) {
  // generic LDS address: low 32 bits are the LDS byte offset (ISA 10.2 aperture map)
  return (unsigned)(uintptr_t)p;
}

// ---------------- WMMA fragment loaders (contiguous k = 16h + s in both A and B) ----------------
// A 16x32 f16: lane half h=lane>>4, row m=lane&15;
//   v0..3 hold K = 16h + {0..7}, v4..7 hold K = 16h + 8 + {0..7} (pairs per VGPR)
__device__ __forceinline__ v16h load_frag_a(const h16* src, int lda, int lane) {
  int hf = lane >> 4, m = lane & 15;
  const h16* row = src + m * lda;
  v16h a;
#pragma unroll
  for (int i = 0; i < 4; ++i) {
    int k0 = hf * 16 + 2 * i;
    a[2 * i]         = row[k0];
    a[2 * i + 1]     = row[k0 + 1];
    int k1 = hf * 16 + 8 + 2 * i;
    a[2 * (i + 4)]     = row[k1];
    a[2 * (i + 4) + 1] = row[k1 + 1];
  }
  return a;
}
// B 32x16 f16 (row-major K x N in LDS): lanes 0-15 hold K=0..15, lanes 16-31 K=16..31;
//   VGPR i holds K = 16h + {2i, 2i+1} for column n = lane&15
__device__ __forceinline__ v16h load_frag_b(const h16* src, int ldb, int lane) {
  int hf = lane >> 4, n = lane & 15;
  v16h b;
#pragma unroll
  for (int i = 0; i < 8; ++i) {
    int k = hf * 16 + 2 * i;
    b[2 * i]     = src[k * ldb + n];
    b[2 * i + 1] = src[(k + 1) * ldb + n];
  }
  return b;
}
// C/D 16x16 f32: VGPR r -> row M = r + 8*(lane>>4), col N = lane&15

// one 16x16 output tile, K = 64 (two 16x16x32 WMMA steps), accumulating into acc
__device__ __forceinline__ v8f gemm_k64(v8f acc, const h16* A, int lda,
                                        const h16* B, int ldb, int lane) {
  v16h a0 = load_frag_a(A, lda, lane);
  v16h b0 = load_frag_b(B, ldb, lane);
  acc = __builtin_amdgcn_wmma_f32_16x16x32_f16(false, a0, false, b0,
                                               (short)0, acc, false, false);
  v16h a1 = load_frag_a(A + 32, lda, lane);
  v16h b1 = load_frag_b(B + 32 * ldb, ldb, lane);
  acc = __builtin_amdgcn_wmma_f32_16x16x32_f16(false, a1, false, b1,
                                               (short)0, acc, false, false);
  return acc;
}

// ---------------- bitonic sort of 4096 uint keys in LDS ----------------
__device__ void bitonic4096(unsigned* keys, int tid, int nthr) {
  for (unsigned k = 2; k <= 4096u; k <<= 1) {
    for (unsigned j = k >> 1; j > 0; j >>= 1) {
      __syncthreads();
      for (unsigned i = tid; i < 4096u; i += nthr) {
        unsigned ixj = i ^ j;
        if (ixj > i) {
          unsigned a = keys[i], b = keys[ixj];
          bool up = ((i & k) == 0);
          if (a != b && ((a > b) == up)) { keys[i] = b; keys[ixj] = a; }
        }
      }
    }
  }
  __syncthreads();
}

// ---------------- K1: LSH hash + sort (one block per bh) ----------------
__global__ void __launch_bounds__(1024)
hash_sort_kernel(const float* __restrict__ q_in, const float* __restrict__ k_in,
                 int* __restrict__ Qsort, int* __restrict__ Ksort) {
  __shared__ float proj[kD * kNPROJ];
  __shared__ unsigned keys[kN];
  int bh = blockIdx.x, tid = threadIdx.x;
  int b = bh >> 3, h = bh & 7;
  for (int i = tid; i < kD * kNPROJ; i += 1024)
    proj[i] = rng_normal(0x00009137u + bh * (kD * kNPROJ) + i);
  __syncthreads();

  // ---- keys for K ----
  for (int n = tid; n < kN; n += 1024) {
    size_t base = (((size_t)b * kN + n) * kH + h) * kD;
    float acc[kNPROJ] = {0.f, 0.f, 0.f, 0.f, 0.f, 0.f, 0.f};
    for (int d = 0; d < kD; ++d) {
      float x = k_in[base + d];
#pragma unroll
      for (int r = 0; r < kNPROJ; ++r) acc[r] += x * proj[d * kNPROJ + r];
    }
    unsigned c = 0;
#pragma unroll
    for (int r = 0; r < kNPROJ; ++r) if (acc[r] > 0.f) c |= (1u << r);
    unsigned gray = c ^ (c >> 1);        // reflected Gray permutation
    keys[n] = (gray << 12) | (unsigned)n; // stable composite key
  }
  bitonic4096(keys, tid, 1024);
  for (int p = tid; p < kN; p += 1024) Ksort[bh * kN + p] = (int)(keys[p] & 0xFFFu);
  __syncthreads();

  // ---- keys for Q (same projections) ----
  for (int n = tid; n < kN; n += 1024) {
    size_t base = (((size_t)b * kN + n) * kH + h) * kD;
    float acc[kNPROJ] = {0.f, 0.f, 0.f, 0.f, 0.f, 0.f, 0.f};
    for (int d = 0; d < kD; ++d) {
      float x = q_in[base + d];
#pragma unroll
      for (int r = 0; r < kNPROJ; ++r) acc[r] += x * proj[d * kNPROJ + r];
    }
    unsigned c = 0;
#pragma unroll
    for (int r = 0; r < kNPROJ; ++r) if (acc[r] > 0.f) c |= (1u << r);
    unsigned gray = c ^ (c >> 1);
    keys[n] = (gray << 12) | (unsigned)n;
  }
  bitonic4096(keys, tid, 1024);
  for (int p = tid; p < kN; p += 1024) Qsort[bh * kN + p] = (int)(keys[p] & 0xFFFu);
}

// ---------------- K2: gather sorted f16 tiles (q16s, k16s, vaug16s) ----------------
__global__ void gather_kernel(const float* __restrict__ q_in, const float* __restrict__ k_in,
                              const float* __restrict__ v_in,
                              const int* __restrict__ Qsort, const int* __restrict__ Ksort,
                              h16* __restrict__ q16s, h16* __restrict__ k16s,
                              h16* __restrict__ vaug) {
  long total_qk = (long)kBH * kN * kD;
  for (long e = (long)blockIdx.x * blockDim.x + threadIdx.x; e < total_qk;
       e += (long)gridDim.x * blockDim.x) {
    int bh = (int)(e / (kN * kD));
    long rem = e - (long)bh * kN * kD;
    int p = (int)(rem >> 6), d = (int)(rem & 63);
    int b = bh >> 3, h = bh & 7;
    int nq = Qsort[bh * kN + p];
    q16s[e] = (h16)q_in[(((size_t)b * kN + nq) * kH + h) * kD + d];
    int nk = Ksort[bh * kN + p];
    k16s[e] = (h16)k_in[(((size_t)b * kN + nk) * kH + h) * kD + d];
  }
  long total_v = (long)kBH * kN * kDVP;
  for (long e = (long)blockIdx.x * blockDim.x + threadIdx.x; e < total_v;
       e += (long)gridDim.x * blockDim.x) {
    int bh = (int)(e / (kN * kDVP));
    long rem = e - (long)bh * kN * kDVP;
    int p = (int)(rem / kDVP), cc = (int)(rem % kDVP);
    int b = bh >> 3, h = bh & 7;
    int nk = Ksort[bh * kN + p];
    float val = (cc < kD) ? v_in[(((size_t)b * kN + nk) * kH + h) * kD + cc]
                          : (cc == kD ? 1.0f : 0.0f);
    vaug[e] = (h16)val;
  }
}

// ---------------- K3: Gram + power iteration + P + CDF (one block per bh) -------
__global__ void __launch_bounds__(256)
gram_power_kernel(const h16* __restrict__ vaug, float* __restrict__ Pglob,
                  float* __restrict__ CDFglob) {
  __shared__ float chunk[64 * kDV];
  __shared__ float gram[kDV * kDV];
  __shared__ float pl[kN];
  __shared__ float xv[kDV], yv[kDV];
  __shared__ float bsum[256];
  __shared__ float snorm, stot;
  int bh = blockIdx.x, tid = threadIdx.x;

  int te[17], de[17]; float g[17]; int cnt = 0;
  for (int e = tid; e < kDV * kDV; e += 256) { te[cnt] = e / kDV; de[cnt] = e % kDV; g[cnt] = 0.f; ++cnt; }

  for (int c = 0; c < kN / 64; ++c) {
    __syncthreads();
    for (int i = tid; i < 64 * kDV; i += 256) {
      int r = i / kDV, col = i - r * kDV;
      chunk[i] = (float)vaug[((size_t)bh * kN + c * 64 + r) * kDVP + col];
    }
    __syncthreads();
    if (tid < 64) {                      // row norms of v_aug (includes the ones column)
      float s = 0.f;
      for (int col = 0; col < kDV; ++col) { float x = chunk[tid * kDV + col]; s += x * x; }
      pl[c * 64 + tid] = sqrtf(s);
    }
    for (int j = 0; j < cnt; ++j) {
      float s = g[j];
      for (int r = 0; r < 64; ++r) s += chunk[r * kDV + te[j]] * chunk[r * kDV + de[j]];
      g[j] = s;
    }
  }
  __syncthreads();
  for (int j = 0; j < cnt; ++j) gram[te[j] * kDV + de[j]] = g[j];
  if (tid < kDV) xv[tid] = rng_normal(0x0051ED27u + bh * kDV + tid);
  __syncthreads();
  if (tid == 0) { float s = 0.f; for (int i = 0; i < kDV; ++i) s += xv[i] * xv[i]; snorm = sqrtf(s); }
  __syncthreads();
  if (tid < kDV) xv[tid] /= snorm;
  __syncthreads();
  for (int it = 0; it < 20; ++it) {
    if (tid < kDV) { float s = 0.f; for (int d2 = 0; d2 < kDV; ++d2) s += gram[tid * kDV + d2] * xv[d2]; yv[tid] = s; }
    __syncthreads();
    if (tid == 0) { float s = 0.f; for (int i = 0; i < kDV; ++i) s += yv[i] * yv[i]; snorm = sqrtf(s); }
    __syncthreads();
    if (tid < kDV) xv[tid] = yv[tid] / snorm;
    __syncthreads();
  }
  float Vn = snorm;
  for (int n = tid; n < kN; n += 256) pl[n] = pl[n] / Vn + (1.0f / (float)kN);
  __syncthreads();
  // two-level inclusive scan of pl[0..4095]
  float r16[16]; float run = 0.f; int base = tid * 16;
  for (int j = 0; j < 16; ++j) { r16[j] = pl[base + j]; run += r16[j]; pl[base + j] = run; }
  bsum[tid] = run;
  __syncthreads();
  if (tid == 0) { float rr = 0.f; for (int i = 0; i < 256; ++i) { float t = bsum[i]; bsum[i] = rr; rr += t; } stot = rr; }
  __syncthreads();
  float off = bsum[tid], tot = stot;
  for (int j = 0; j < 16; ++j) {
    int n = base + j;
    Pglob[(size_t)bh * kN + n]   = r16[j] / tot;
    CDFglob[(size_t)bh * kN + n] = (pl[n] + off) / tot;
  }
}

// ---------------- K4: categorical sampling + build Kpi / sigma*Vpi ----------------
__global__ void __launch_bounds__(256)
sample_kernel(const float* __restrict__ Pglob, const float* __restrict__ CDF,
              const h16* __restrict__ k16s, const h16* __restrict__ vaug,
              int* __restrict__ sblk, h16* __restrict__ Kpi, h16* __restrict__ Vsig) {
  int bh = blockIdx.x, tid = threadIdx.x;
  const float* cdf = CDF + (size_t)bh * kN;
  for (int t = tid; t < kSPAD; t += 256) {
    size_t kout = ((size_t)bh * kSPAD + t) * kD;
    size_t vout = ((size_t)bh * kSPAD + t) * kDVP;
    if (t < kSAMPLE) {
      unsigned hsh = pcg(pcg(0x0A5E11CEu + bh * kSAMPLE + t));
      float u = ((float)hsh + 0.5f) * (1.0f / 4294967296.0f);
      int lo = 0, hi = kN - 1;
      while (lo < hi) { int mid = (lo + hi) >> 1; if (cdf[mid] < u) lo = mid + 1; else hi = mid; }
      int s = lo;
      sblk[bh * kSPAD + t] = s >> 6;   // sample's sorted-key block id
      float sig = 1.0f / (Pglob[(size_t)bh * kN + s] * (float)kSAMPLE);
      const h16* krow = k16s + ((size_t)bh * kN + s) * kD;
      for (int d = 0; d < kD; ++d) Kpi[kout + d] = krow[d];
      const h16* vrow = vaug + ((size_t)bh * kN + s) * kDVP;
      for (int c = 0; c < kDVP; ++c)
        Vsig[vout + c] = (c < kDV) ? (h16)((float)vrow[c] * sig) : (h16)0.f;
    } else {  // padded samples contribute exactly zero
      sblk[bh * kSPAD + t] = -1;
      for (int d = 0; d < kD; ++d) Kpi[kout + d] = (h16)0.f;
      for (int c = 0; c < kDVP; ++c) Vsig[vout + c] = (h16)0.f;
    }
  }
}

// ---------------- K5: block-sparse attention (WMMA + TDM), sorted-q order -------
__global__ void __launch_bounds__(256)
sparse_attn_kernel(const h16* __restrict__ q16s, const h16* __restrict__ k16s,
                   const h16* __restrict__ vaug, float* __restrict__ attS) {
  __shared__ __align__(16) h16 Aq[64 * 64];
  __shared__ __align__(16) h16 Kt[64 * 66];     // transposed keys, padded stride
  __shared__ __align__(16) h16 Bw[64 * kDVP];
  __shared__ __align__(16) h16 S16[64 * 64];
  int bid = blockIdx.x, bh = bid >> 6, blk = bid & 63;
  int tid = threadIdx.x, wave = tid >> 5, lane = tid & 31;
  int hf = lane >> 4, nl = lane & 15;
  size_t rowbase = (size_t)bh * kN + blk * 64;

  // TDM: DMA the two contiguous tiles into LDS while threads do the K transpose
  if (wave == 0) {
    tdm_load_1d(lds_off(Aq), q16s + rowbase * kD, 64 * kD / 4);        // 8192B
    tdm_load_1d(lds_off(Bw), vaug + rowbase * kDVP, 64 * kDVP / 4);    // 10240B
  }
  for (int i = tid; i < 64 * 64; i += 256) {
    int p = i >> 6, d = i & 63;
    Kt[d * 66 + p] = k16s[rowbase * kD + i];
  }
  if (wave == 0) __builtin_amdgcn_s_wait_tensorcnt(0);
  __syncthreads();

  // GEMM1: S = exp(Qs Ks^T)   (16 tiles over 8 waves)
  for (int t = wave; t < 16; t += 8) {
    int mt = t >> 2, nt = t & 3;
    v8f acc = {};
    acc = gemm_k64(acc, Aq + mt * 16 * 64, 64, Kt + nt * 16, 66, lane);
#pragma unroll
    for (int r = 0; r < 8; ++r) {
      float e = __expf(acc[r]);
      S16[(mt * 16 + hf * 8 + r) * 64 + nt * 16 + nl] = (h16)e;
    }
  }
  __syncthreads();

  // GEMM2: attS = S x [V|1]   (20 tiles over 8 waves; cols 65..79 are zero pad)
  for (int t = wave; t < 20; t += 8) {
    int mt = t / 5, nt = t % 5;
    v8f acc = {};
    acc = gemm_k64(acc, S16 + mt * 16 * 64, 64, Bw + nt * 16, kDVP, lane);
#pragma unroll
    for (int r = 0; r < 8; ++r) {
      int row = mt * 16 + hf * 8 + r;
      int col = nt * 16 + nl;
      if (col < kDV) attS[(rowbase + row) * kATTLD + col] = acc[r];
    }
  }
}

// ---------------- K6: residual (WMMA + TDM, persistent accumulators) ------------
__global__ void __launch_bounds__(256)
residual_kernel(const h16* __restrict__ q16s, const h16* __restrict__ Kpi,
                const h16* __restrict__ Vsig, const int* __restrict__ sblk,
                const float* __restrict__ attS, const int* __restrict__ Qsort,
                float* __restrict__ out) {
  __shared__ __align__(16) h16 Aq[64 * 64];
  __shared__ __align__(16) h16 Kt[64 * 66];
  __shared__ __align__(16) h16 Vs[64 * kDVP];
  __shared__ __align__(16) h16 S16[64 * 64];
  __shared__ __align__(16) float attb[64 * kDVP];
  __shared__ int sb[64];
  int bid = blockIdx.x, bh = bid >> 6, blk = bid & 63;
  int tid = threadIdx.x, wave = tid >> 5, lane = tid & 31;
  int hf = lane >> 4, nl = lane & 15;
  size_t rowbase = (size_t)bh * kN + blk * 64;

  if (wave == 0) tdm_load_1d(lds_off(Aq), q16s + rowbase * kD, 64 * kD / 4);

  v8f acc0 = {}, acc1 = {}, acc2 = {};
  for (int c = 0; c < kSPAD / 64; ++c) {
    __syncthreads();
    size_t kbase = ((size_t)bh * kSPAD + c * 64) * kD;
    size_t vbase = ((size_t)bh * kSPAD + c * 64) * kDVP;
    if (wave == 0) tdm_load_1d(lds_off(Vs), Vsig + vbase, 64 * kDVP / 4);
    for (int i = tid; i < 64 * 64; i += 256) {
      int s = i >> 6, d = i & 63;
      Kt[d * 66 + s] = Kpi[kbase + i];
    }
    if (tid < 64) sb[tid] = sblk[bh * kSPAD + c * 64 + tid];
    if (c + 1 < kSPAD / 64) __builtin_prefetch(&Kpi[kbase + 64 * kD], 0, 1);
    if (wave == 0) __builtin_amdgcn_s_wait_tensorcnt(0);
    __syncthreads();

    // GEMM1: S = exp(Q Kpi^T) * mask  (mask is per-column: sample in our block)
    for (int t = wave; t < 16; t += 8) {
      int mt = t >> 2, nt = t & 3;
      v8f a = {};
      a = gemm_k64(a, Aq + mt * 16 * 64, 64, Kt + nt * 16, 66, lane);
      float msk = (sb[nt * 16 + nl] == blk) ? 0.f : 1.f;
#pragma unroll
      for (int r = 0; r < 8; ++r) {
        float e = __expf(a[r]) * msk;
        S16[(mt * 16 + hf * 8 + r) * 64 + nt * 16 + nl] = (h16)e;
      }
    }
    __syncthreads();

    // GEMM2: accumulate S x (sigma * Vpi) with persistent f32 accumulators
    { int t = wave;      int mt = t / 5, nt = t % 5; acc0 = gemm_k64(acc0, S16 + mt * 16 * 64, 64, Vs + nt * 16, kDVP, lane); }
    { int t = wave + 8;  int mt = t / 5, nt = t % 5; acc1 = gemm_k64(acc1, S16 + mt * 16 * 64, 64, Vs + nt * 16, kDVP, lane); }
    if (wave < 4) { int t = wave + 16; int mt = t / 5, nt = t % 5; acc2 = gemm_k64(acc2, S16 + mt * 16 * 64, 64, Vs + nt * 16, kDVP, lane); }
  }
  __syncthreads();
  { int t = wave;     int mt = t / 5, nt = t % 5;
#pragma unroll
    for (int r = 0; r < 8; ++r) attb[(mt * 16 + hf * 8 + r) * kDVP + nt * 16 + nl] = acc0[r]; }
  { int t = wave + 8; int mt = t / 5, nt = t % 5;
#pragma unroll
    for (int r = 0; r < 8; ++r) attb[(mt * 16 + hf * 8 + r) * kDVP + nt * 16 + nl] = acc1[r]; }
  if (wave < 4) { int t = wave + 16; int mt = t / 5, nt = t % 5;
#pragma unroll
    for (int r = 0; r < 8; ++r) attb[(mt * 16 + hf * 8 + r) * kDVP + nt * 16 + nl] = acc2[r]; }
  __syncthreads();

  // att_final = att_sparse + att_res; normalize by column 64; scatter via Qsort
  for (int i = tid; i < 64 * 64; i += 256) {
    int p = i >> 6, d = i & 63;
    float num = attb[p * kDVP + d]  + attS[(rowbase + p) * kATTLD + d];
    float den = attb[p * kDVP + 64] + attS[(rowbase + p) * kATTLD + 64];
    int n = Qsort[rowbase + p];
    out[((size_t)bh * kN + n) * kD + d] = num / den;
  }
}

// ---------------- launch ----------------
extern "C" void kernel_launch(void* const* d_in, const int* in_sizes, int n_in,
                              void* d_out, int out_size, void* d_ws, size_t ws_size,
                              hipStream_t stream) {
  const float* q_in = (const float*)d_in[0];
  const float* k_in = (const float*)d_in[1];
  const float* v_in = (const float*)d_in[2];
  float* out = (float*)d_out;

  char* ws = (char*)d_ws;
  size_t off = 0;
  auto alloc = [&](size_t bytes) -> char* {
    char* p = ws + off;
    off += (bytes + 255) & ~(size_t)255;
    return p;
  };
  int*  Qsort = (int*)alloc((size_t)kBH * kN * sizeof(int));
  int*  Ksort = (int*)alloc((size_t)kBH * kN * sizeof(int));
  h16*  q16s  = (h16*)alloc((size_t)kBH * kN * kD * sizeof(h16));
  h16*  k16s  = (h16*)alloc((size_t)kBH * kN * kD * sizeof(h16));
  h16*  vaug  = (h16*)alloc((size_t)kBH * kN * kDVP * sizeof(h16));
  float* attS = (float*)alloc((size_t)kBH * kN * kATTLD * sizeof(float));
  float* Pg   = (float*)alloc((size_t)kBH * kN * sizeof(float));
  float* CDFg = (float*)alloc((size_t)kBH * kN * sizeof(float));
  int*  sblk  = (int*)alloc((size_t)kBH * kSPAD * sizeof(int));
  h16*  Kpi   = (h16*)alloc((size_t)kBH * kSPAD * kD * sizeof(h16));
  h16*  Vsig  = (h16*)alloc((size_t)kBH * kSPAD * kDVP * sizeof(h16));
  if (off > ws_size) return;  // insufficient scratch

  hash_sort_kernel<<<kBH, 1024, 0, stream>>>(q_in, k_in, Qsort, Ksort);
  gather_kernel<<<2048, 256, 0, stream>>>(q_in, k_in, v_in, Qsort, Ksort, q16s, k16s, vaug);
  gram_power_kernel<<<kBH, 256, 0, stream>>>(vaug, Pg, CDFg);
  sample_kernel<<<kBH, 256, 0, stream>>>(Pg, CDFg, k16s, vaug, sblk, Kpi, Vsig);
  sparse_attn_kernel<<<kBH * kNBLK, 256, 0, stream>>>(q16s, k16s, vaug, attS);
  residual_kernel<<<kBH * kNBLK, 256, 0, stream>>>(q16s, Kpi, Vsig, sblk, attS, Qsort, out);
}
